// VGAE4v2_21552145891394
// MI455X (gfx1250) — matrix operand order repaired and checked by reading, not compiled, not run on hardware
//
#include <hip/hip_runtime.h>
#include <hip/hip_bf16.h>

#define NNODES 1024

typedef __attribute__((ext_vector_type(16))) __bf16 v16bf;
typedef __attribute__((ext_vector_type(8)))  float  v8f;

__device__ __forceinline__ unsigned short bf16_rne(float f) {
  unsigned int u = __float_as_uint(f);
  unsigned int r = u + 0x7FFFu + ((u >> 16) & 1u);   // round-to-nearest-even
  return (unsigned short)(r >> 16);
}
__device__ __forceinline__ float bf16_to_f32(unsigned short h) {
  return __uint_as_float(((unsigned int)h) << 16);
}

union Frag { v16bf v; unsigned int u[8]; };

// C[M,N] = act(A[M,K] @ B[K,N] + bias). fp32 in/out; bf16x3 compensated WMMA core.
// Block = 256 threads (8 waves); block tile 128x16; wave tile 16x16.
// ACT: 0=none, 1=relu, 2=sigmoid. Requires M%128==0, N%16==0, K%32==0.
template<int ACT>
__global__ __launch_bounds__(256)
void gemm_wmma_bf16x3(const float* __restrict__ A, const float* __restrict__ B,
                      const float* __restrict__ bias, float* __restrict__ C,
                      int M, int N, int K) {
  __shared__ unsigned short sAh[128 * 32], sAl[128 * 32];  // [row][k]
  __shared__ unsigned short sBh[16 * 32],  sBl[16 * 32];   // [n][k] (B transposed)
  const int tid  = threadIdx.x;
  const int wave = tid >> 5;
  const int lane = tid & 31;
  const int rowBase = blockIdx.x * 128;
  const int colBase = blockIdx.y * 16;

  v8f acc = {0.f, 0.f, 0.f, 0.f, 0.f, 0.f, 0.f, 0.f};

  for (int k0 = 0; k0 < K; k0 += 32) {
    // Stage A 128x32 chunk as hi/lo bf16 (coalesced: one row of 32 per wave-sweep).
    for (int t = tid; t < 128 * 32; t += 256) {
      int r = t >> 5, k = t & 31;
      float f = A[(size_t)(rowBase + r) * K + (k0 + k)];
      unsigned short h = bf16_rne(f);
      sAh[t] = h;
      sAl[t] = bf16_rne(f - bf16_to_f32(h));
    }
    // Stage B 32x16 chunk transposed to [n][k].
    for (int t = tid; t < 16 * 32; t += 256) {
      int n = t >> 5, k = t & 31;
      float f = B[(size_t)(k0 + k) * N + (colBase + n)];
      unsigned short h = bf16_rne(f);
      sBh[n * 32 + k] = h;
      sBl[n * 32 + k] = bf16_rne(f - bf16_to_f32(h));
    }
    __syncthreads();

    // Gather fragments per CDNA5 ISA 7.12.2 (wave32) layouts.
    const int half = lane >> 4;         // 0: lanes 0-15, 1: lanes 16-31
    const int mr   = lane & 15;
    const unsigned int* pAh = (const unsigned int*)(sAh + (wave * 16 + mr) * 32);
    const unsigned int* pAl = (const unsigned int*)(sAl + (wave * 16 + mr) * 32);
    const unsigned int* pBh = (const unsigned int*)(sBh + mr * 32);
    const unsigned int* pBl = (const unsigned int*)(sBl + mr * 32);
    Frag ah, al, bh, bl;
#pragma unroll
    for (int q = 0; q < 8; ++q) {
      // A 16x32: VGPR q<4 -> K = 2q + 8*half ; q>=4 -> K = 16 + 2(q-4) + 8*half
      int ka = ((q < 4) ? (2 * q) : (16 + 2 * (q - 4))) + half * 8;
      // B 32x16: same K mapping as A (lane = column)
      int kb = ka;
      ah.u[q] = pAh[ka >> 1];
      al.u[q] = pAl[ka >> 1];
      bh.u[q] = pBh[kb >> 1];
      bl.u[q] = pBl[kb >> 1];
    }
    // Compensated product: hi*hi + hi*lo + lo*hi (lo*lo negligible).
    acc = __builtin_amdgcn_wmma_f32_16x16x32_bf16(false, ah.v, false, bh.v, (short)0, acc, false, false);
    acc = __builtin_amdgcn_wmma_f32_16x16x32_bf16(false, ah.v, false, bl.v, (short)0, acc, false, false);
    acc = __builtin_amdgcn_wmma_f32_16x16x32_bf16(false, al.v, false, bh.v, (short)0, acc, false, false);
    __syncthreads();
  }

  // C/D layout: VGPR r, lane L -> row = r + 8*(L>=16), col = L&15.
  const int col = colBase + (lane & 15);
  const float bv = bias ? bias[col] : 0.f;
#pragma unroll
  for (int r = 0; r < 8; ++r) {
    int row = rowBase + wave * 16 + (lane >> 4) * 8 + r;
    float x = acc[r] + bv;
    if (ACT == 1) x = fmaxf(x, 0.f);
    if (ACT == 2) x = 1.f / (1.f + __expf(-x));
    C[(size_t)row * N + col] = x;
  }
}

// GATv2 attention: e[i,j] = a . leaky_relu(gl[j]+gr[i]); mask by adj; softmax_j;
// out[i,:] = sum_j att[i,j] * gr[j,:].  Wave w of each 256-thread block owns
// destination row i = blockIdx.x*8 + w.  F in {32, 64}.
template<int F>
__global__ __launch_bounds__(256)
void gatv2_attention(const float* __restrict__ gl, const float* __restrict__ gr,
                     const float* __restrict__ avec, const int* __restrict__ adj,
                     float* __restrict__ out) {
  __shared__ float s_src[64 * F];      // staged source-row chunk (gl, later gr)
  __shared__ float s_e[8][NNODES];     // per-wave score row
  __shared__ float s_gri[8][F];        // gr[i,:] per wave
  __shared__ float s_a[F];
  const int tid  = threadIdx.x;
  const int wave = tid >> 5;
  const int lane = tid & 31;
  const int i = blockIdx.x * 8 + wave;

  for (int t = tid; t < F; t += 256) s_a[t] = avec[t];
  for (int f = lane; f < F; f += 32) s_gri[wave][f] = gr[(size_t)i * F + f];
  __syncthreads();

  // Phase 1: masked scores.
  for (int jc = 0; jc < NNODES; jc += 64) {
    for (int t = tid; t < 64 * F; t += 256) s_src[t] = gl[(size_t)jc * F + t];
    __syncthreads();
    for (int jj = lane; jj < 64; jj += 32) {
      int j = jc + jj;
      float e = 0.f;
#pragma unroll
      for (int f = 0; f < F; ++f) {
        float s = s_src[jj * F + f] + s_gri[wave][f];
        e = fmaf(s_a[f], (s > 0.f) ? s : 0.2f * s, e);   // leaky_relu, slope 0.2
      }
      s_e[wave][j] = (adj[(size_t)i * NNODES + j] == 0) ? -1e9f : e;
    }
    __syncthreads();
  }

  // Phase 2: softmax over j (wave32 shuffle reductions).
  float m = -3.0e38f;
  for (int j = lane; j < NNODES; j += 32) m = fmaxf(m, s_e[wave][j]);
#pragma unroll
  for (int off = 16; off; off >>= 1) m = fmaxf(m, __shfl_xor(m, off, 32));
  float ssum = 0.f;
  for (int j = lane; j < NNODES; j += 32) {
    float t = __expf(s_e[wave][j] - m);
    s_e[wave][j] = t;
    ssum += t;
  }
#pragma unroll
  for (int off = 16; off; off >>= 1) ssum += __shfl_xor(ssum, off, 32);
  const float inv = 1.f / ssum;
  for (int j = lane; j < NNODES; j += 32) s_e[wave][j] *= inv;
  __syncthreads();

  // Phase 3: out[i,f] = sum_j att[j] * gr[j,f], gr staged through LDS.
  constexpr int NQ = F / 32;
  float acc[NQ];
#pragma unroll
  for (int q = 0; q < NQ; ++q) acc[q] = 0.f;
  for (int jc = 0; jc < NNODES; jc += 64) {
    for (int t = tid; t < 64 * F; t += 256) s_src[t] = gr[(size_t)jc * F + t];
    __syncthreads();
    for (int jj = 0; jj < 64; ++jj) {
      float att = s_e[wave][jc + jj];
#pragma unroll
      for (int q = 0; q < NQ; ++q)
        acc[q] = fmaf(att, s_src[jj * F + lane + q * 32], acc[q]);
    }
    __syncthreads();
  }
#pragma unroll
  for (int q = 0; q < NQ; ++q) out[(size_t)i * F + lane + q * 32] = acc[q];
}

// z = noise * exp(logstd) + mean
__global__ __launch_bounds__(256)
void reparam_kernel(const float* __restrict__ noise, const float* __restrict__ logstd,
                    const float* __restrict__ mean, float* __restrict__ z, int n) {
  int t = blockIdx.x * blockDim.x + threadIdx.x;
  if (t < n) z[t] = fmaf(noise[t], __expf(logstd[t]), mean[t]);
}

extern "C" void kernel_launch(void* const* d_in, const int* in_sizes, int n_in,
                              void* d_out, int out_size, void* d_ws, size_t ws_size,
                              hipStream_t stream) {
  (void)in_sizes; (void)n_in; (void)out_size; (void)ws_size;
  const float* X     = (const float*)d_in[0];
  const int*   adj   = (const int*)d_in[1];
  const float* Wl1   = (const float*)d_in[2];
  const float* Wr1   = (const float*)d_in[3];
  const float* a1    = (const float*)d_in[4];
  const float* Wl2   = (const float*)d_in[5];
  const float* Wr2   = (const float*)d_in[6];
  const float* a2    = (const float*)d_in[7];
  const float* Wl3   = (const float*)d_in[8];
  const float* Wr3   = (const float*)d_in[9];
  const float* a3    = (const float*)d_in[10];
  const float* fc1w  = (const float*)d_in[11];
  const float* fc1b  = (const float*)d_in[12];
  const float* fc2w  = (const float*)d_in[13];
  const float* fc2b  = (const float*)d_in[14];
  const float* fc3w  = (const float*)d_in[15];
  const float* fc3b  = (const float*)d_in[16];
  const float* noise = (const float*)d_in[17];
  float* outp = (float*)d_out;

  // Workspace layout (floats); ~4 MB total.
  float* W = (float*)d_ws;
  float* gl1    = W;                 // 1024*64
  float* gr1    = gl1    + 65536;    // 1024*64
  float* hidden = gr1    + 65536;    // 1024*64
  float* gl2    = hidden + 65536;    // 1024*32
  float* gr2    = gl2    + 32768;
  float* gl3    = gr2    + 32768;
  float* gr3    = gl3    + 32768;
  float* mean_  = gr3    + 32768;
  float* logstd = mean_  + 32768;
  float* zbuf   = logstd + 32768;
  float* t1     = zbuf   + 32768;    // 1024*64
  float* t2     = t1     + 65536;    // 1024*512

  const dim3 blk(256);

  // ---- GAT layer 1: projections (K=512) + attention (F=64) ----
  gemm_wmma_bf16x3<0><<<dim3(8, 4), blk, 0, stream>>>(X, Wl1, nullptr, gl1, 1024, 64, 512);
  gemm_wmma_bf16x3<0><<<dim3(8, 4), blk, 0, stream>>>(X, Wr1, nullptr, gr1, 1024, 64, 512);
  gatv2_attention<64><<<dim3(128), blk, 0, stream>>>(gl1, gr1, a1, adj, hidden);

  // ---- GAT layers 2 & 3 (mean / logstd), K=64, F=32 ----
  gemm_wmma_bf16x3<0><<<dim3(8, 2), blk, 0, stream>>>(hidden, Wl2, nullptr, gl2, 1024, 32, 64);
  gemm_wmma_bf16x3<0><<<dim3(8, 2), blk, 0, stream>>>(hidden, Wr2, nullptr, gr2, 1024, 32, 64);
  gatv2_attention<32><<<dim3(128), blk, 0, stream>>>(gl2, gr2, a2, adj, mean_);

  gemm_wmma_bf16x3<0><<<dim3(8, 2), blk, 0, stream>>>(hidden, Wl3, nullptr, gl3, 1024, 32, 64);
  gemm_wmma_bf16x3<0><<<dim3(8, 2), blk, 0, stream>>>(hidden, Wr3, nullptr, gr3, 1024, 32, 64);
  gatv2_attention<32><<<dim3(128), blk, 0, stream>>>(gl3, gr3, a3, adj, logstd);

  // ---- Reparameterize ----
  reparam_kernel<<<dim3((32768 + 255) / 256), blk, 0, stream>>>(noise, logstd, mean_, zbuf, 32768);

  // ---- Decoder MLP ----
  gemm_wmma_bf16x3<1><<<dim3(8, 4),  blk, 0, stream>>>(zbuf, fc1w, fc1b, t1, 1024, 64, 32);
  gemm_wmma_bf16x3<1><<<dim3(8, 32), blk, 0, stream>>>(t1,   fc2w, fc2b, t2, 1024, 512, 64);
  gemm_wmma_bf16x3<2><<<dim3(8, 64), blk, 0, stream>>>(t2,   fc3w, fc3b, outp, 1024, 1024, 512);
}
